// Block_35983236006219
// MI455X (gfx1250) — matrix operand order repaired and checked by reading, not compiled, not run on hardware
//
#include <hip/hip_runtime.h>

typedef _Float16 v16h __attribute__((ext_vector_type(16)));
typedef _Float16 v8h  __attribute__((ext_vector_type(8)));
typedef float    v8f  __attribute__((ext_vector_type(8)));
typedef unsigned int su4 __attribute__((ext_vector_type(4)));
typedef unsigned int su8 __attribute__((ext_vector_type(8)));

#define B_    512
#define N_    196
#define C_    512
#define NH_   8
#define HD_   64
#define AG_   49
#define AGP_  64
#define NPAD_ 224   // 196 padded to 14*16 (multiple of 32 for WMMA K)

// ---------------- WMMA helpers (CDNA5 16x16x32 f16 -> f32) ----------------

static __device__ __forceinline__ v16h ldfrag(const _Float16* p0, const _Float16* p1) {
  v8h a = *(const v8h*)p0;
  v8h b = *(const v8h*)p1;
  v16h r;
#pragma unroll
  for (int i = 0; i < 8; ++i) { r[i] = a[i]; r[i + 8] = b[i]; }
  return r;
}

static __device__ __forceinline__ v8f wmma_f16(v16h a, v16h b, v8f c) {
  return __builtin_amdgcn_wmma_f32_16x16x32_f16(false, a, false, b, (short)0, c, false, false);
}

// ---------------- TDM: 2D tile (16-bit elements) global -> LDS ----------------
// D# group0/group1 packed per CDNA5 ISA 8.3/8.4; count=1, type=2 ("image"),
// data_size=1 (2 bytes). Rows past tensor_dim1 are OOB (zero / dropped) --
// the consumers mask those pad rows after WMMA, so either behavior is safe.
static __device__ __forceinline__ void tdm_load_2d_f16(const void* gptr, unsigned lds_byte_off,
                                                       unsigned width, unsigned rows,
                                                       unsigned tile_rows, unsigned row_stride) {
  unsigned long long ga = (unsigned long long)(uintptr_t)gptr;
  su4 g0;
  g0[0] = 1u;                                            // count=1, no gather
  g0[1] = lds_byte_off;                                  // lds_addr (bytes)
  g0[2] = (unsigned)(ga & 0xffffffffu);                  // global_addr lo
  g0[3] = (unsigned)((ga >> 32) & 0x01ffffffu) | 0x80000000u;  // addr hi | type=2
  su8 g1;
  g1[0] = 1u << 16;                                      // data_size = 1 (2B)
  g1[1] = (width & 0xffffu) << 16;                       // tensor_dim0[15:0]
  g1[2] = (width >> 16) | ((rows & 0xffffu) << 16);      // dim0 hi | tensor_dim1 lo
  g1[3] = (rows >> 16) | ((width & 0xffffu) << 16);      // dim1 hi | tile_dim0
  g1[4] = tile_rows & 0xffffu;                           // tile_dim1 (tile_dim2=0)
  g1[5] = row_stride;                                    // tensor_dim0_stride lo
  g1[6] = 0u;                                            // stride hi | dim1_stride lo
  g1[7] = 0u;
  asm volatile("tensor_load_to_lds %0, %1" :: "s"(g0), "s"(g1) : "memory");
}

// ---------------- prep kernels ----------------

__global__ void cvt_f32_to_f16(const float* __restrict__ in, _Float16* __restrict__ out, int n) {
  int i = blockIdx.x * 256 + threadIdx.x;
  if (i < n) out[i] = (_Float16)in[i];
}

// w: (N x K) row-major -> Wf[(K/32)][N][32] (fragment-native B layout)
__global__ void repack_weight(const float* __restrict__ w, _Float16* __restrict__ wf, int N, int K) {
  int i = blockIdx.x * 256 + threadIdx.x;
  if (i < N * K) {
    int j = i / K, k = i - j * K;
    wf[((long)(k >> 5) * N + j) * 32 + (k & 31)] = (_Float16)w[i];
  }
}

static __device__ __forceinline__ float bilerp7(const float* __restrict__ p, int oy, int ox) {
  float sy = oy * 0.5f - 0.25f;
  float sx = ox * 0.5f - 0.25f;
  int y0 = (int)floorf(sy); float fy = sy - (float)y0;
  int x0 = (int)floorf(sx); float fx = sx - (float)x0;
  int y1 = y0 + 1, x1 = x0 + 1;
  y0 = y0 < 0 ? 0 : (y0 > 6 ? 6 : y0);
  y1 = y1 < 0 ? 0 : (y1 > 6 ? 6 : y1);
  x0 = x0 < 0 ? 0 : (x0 > 6 ? 6 : x0);
  x1 = x1 < 0 ? 0 : (x1 > 6 ? 6 : x1);
  float a = (1.f - fx) * p[y0 * 7 + x0] + fx * p[y0 * 7 + x1];
  float b = (1.f - fx) * p[y1 * 7 + x0] + fx * p[y1 * 7 + x1];
  return (1.f - fy) * a + fy * b;
}

__global__ void bias_kernel(const float* __restrict__ an, const float* __restrict__ na,
                            const float* __restrict__ ahb, const float* __restrict__ awb,
                            const float* __restrict__ hab, const float* __restrict__ wab,
                            float* __restrict__ bias1, float* __restrict__ bias2) {
  int idx = blockIdx.x * 256 + threadIdx.x;
  const int total = NH_ * AG_ * N_;
  if (idx < total) {                       // bias1[h][a][n]
    int n = idx % N_;
    int a = (idx / N_) % AG_;
    int h = idx / (N_ * AG_);
    int oy = n / 14, ox = n % 14;
    float r = bilerp7(an + (h * AG_ + a) * 49, oy, ox);
    bias1[idx] = r + ahb[(h * AG_ + a) * 14 + oy] + awb[(h * AG_ + a) * 14 + ox];
  } else if (idx < 2 * total) {            // bias2[h][n][a]
    int i2 = idx - total;
    int a = i2 % AG_;
    int n = (i2 / AG_) % N_;
    int h = i2 / (AG_ * N_);
    int oy = n / 14, ox = n % 14;
    float r = bilerp7(na + (h * AG_ + a) * 49, oy, ox);
    bias2[i2] = r + hab[(h * 14 + oy) * AG_ + a] + wab[(h * 14 + ox) * AG_ + a];
  }
}

// ---------------- generic WMMA GEMM: out = A(MxK,f16) * W^T + bias ----------------
// Wf in fragment layout [(K/32)][N][32]. Block = 8 waves covering 128x128;
// each wave owns a 16(M) x 128(N) strip -> 8 WMMAs per A-fragment (high reuse).
template <bool OUT_F16>
__global__ __launch_bounds__(256) void gemm_wmma(const _Float16* __restrict__ A,
                                                 const _Float16* __restrict__ Wf,
                                                 const float* __restrict__ bias,
                                                 void* __restrict__ outp,
                                                 int M, int K, int N) {
  const int lane = threadIdx.x & 31;
  const int wv   = threadIdx.x >> 5;
  const long m0  = (long)blockIdx.y * 128 + wv * 16;
  const int  n0  = blockIdx.x * 128;

  v8f z = {0,0,0,0,0,0,0,0};
  v8f acc[8];
#pragma unroll
  for (int j = 0; j < 8; ++j) acc[j] = z;

  const _Float16* arow = A + (m0 + (lane & 15)) * (long)K + ((lane >> 4) * 8);
  for (int k0 = 0; k0 < K; k0 += 32) {
    v16h af = ldfrag(arow + k0, arow + k0 + 16);
    const _Float16* bb = Wf + ((long)(k0 >> 5) * N + n0 + (lane & 15)) * 32 + ((lane >> 4) * 8);
#pragma unroll
    for (int j = 0; j < 8; ++j)
      acc[j] = wmma_f16(af, ldfrag(bb + j * 512, bb + j * 512 + 16), acc[j]);
  }

#pragma unroll
  for (int j = 0; j < 8; ++j) {
    int n = n0 + j * 16 + (lane & 15);
    float bval = bias[n];
#pragma unroll
    for (int r = 0; r < 8; ++r) {
      long m = m0 + r + ((lane >> 4) * 8);
      float v = acc[j][r] + bval;
      if (OUT_F16) ((_Float16*)outp)[m * N + n] = (_Float16)v;
      else         ((float*)outp)[m * N + n] = v;
    }
  }
}

// ---------------- fused agent-attention kernel: one workgroup per (b, h) ----------------
__global__ __launch_bounds__(256) void attn_kernel(const _Float16* __restrict__ qkv,
                                                   const float* __restrict__ bias1,
                                                   const float* __restrict__ bias2,
                                                   const float* __restrict__ dwcw,
                                                   const float* __restrict__ dwcb,
                                                   _Float16* __restrict__ aout) {
  __shared__ __align__(64) _Float16 q_s[NPAD_ * HD_];   // [n][d]  (TDM loaded)
  __shared__ __align__(64) _Float16 k_s[NPAD_ * HD_];   // [n][d]  (TDM loaded)
  __shared__ __align__(64) _Float16 vT [HD_ * NPAD_];   // [d][n]  (manual transpose, zero pad)
  __shared__ __align__(64) _Float16 ag [AGP_ * HD_];    // [a][d]
  __shared__ __align__(64) _Float16 agvT[HD_ * AGP_];   // [d][a]
  __shared__ __align__(64) _Float16 sc [AGP_ * NPAD_];  // scores (s1: [a][224], s2: [n][64])
  __shared__ __align__(64) _Float16 at [AGP_ * NPAD_];  // attn   (s1: [a][224], s2: [n][64])

  const int tid = threadIdx.x, lane = tid & 31, wv = tid >> 5;
  const int b = blockIdx.x >> 3, h = blockIdx.x & 7;
  const _Float16* base = qkv + (long)b * N_ * (3 * C_) + h * HD_;

  // ---- wave 0 kicks off TDM DMA of the q and k tiles (196x64 halfs, stride 1536)
  if (wv == 0) {
    tdm_load_2d_f16(base,       (unsigned)(uintptr_t)q_s, HD_, N_, NPAD_, 3 * C_);
    tdm_load_2d_f16(base + C_,  (unsigned)(uintptr_t)k_s, HD_, N_, NPAD_, 3 * C_);
  }

  // ---- meanwhile all waves stream v and transpose into vT (explicit zero pad)
  for (int idx = tid; idx < NPAD_ * 8; idx += 256) {
    int n  = idx >> 3;
    int c8 = (idx & 7) * 8;
    v8h val;
    if (n < N_) {
      val = *(const v8h*)(base + (long)n * (3 * C_) + 2 * C_ + c8);
    } else {
#pragma unroll
      for (int i = 0; i < 8; ++i) val[i] = (_Float16)0.f;
    }
#pragma unroll
    for (int i = 0; i < 8; ++i) vT[(c8 + i) * NPAD_ + n] = val[i];
  }

  if (wv == 0) __builtin_amdgcn_s_wait_tensorcnt(0);
  __syncthreads();

  // ---- agent tokens: 2x2 mean pool of q over the 14x14 grid (rows >=49 zeroed)
  for (int idx = tid; idx < AGP_ * HD_; idx += 256) {
    int a = idx >> 6, d = idx & 63;
    _Float16 r = (_Float16)0.f;
    if (a < AG_) {
      int ay = a / 7, ax = a % 7;
      int n00 = (2 * ay) * 14 + 2 * ax;
      float s = (float)q_s[n00 * HD_ + d] + (float)q_s[(n00 + 1) * HD_ + d] +
                (float)q_s[(n00 + 14) * HD_ + d] + (float)q_s[(n00 + 15) * HD_ + d];
      r = (_Float16)(0.25f * s);
    }
    ag[idx] = r;
  }
  __syncthreads();

  // ---- stage 1 scores: ag(64x64) . k^T -> sc[a][n] (64x224), scale + bias + masks
  for (int t = wv; t < 56; t += 8) {
    int tm = t / 14, tn = t - tm * 14;
    v8f acc = {0,0,0,0,0,0,0,0};
#pragma unroll
    for (int k0 = 0; k0 < HD_; k0 += 32) {
      const _Float16* ap = ag + (tm * 16 + (lane & 15)) * HD_ + k0 + ((lane >> 4) * 8);
      v16h af = ldfrag(ap, ap + 16);
      const _Float16* bp = k_s + (tn * 16 + (lane & 15)) * HD_ + k0 + ((lane >> 4) * 8);
      v16h bf = ldfrag(bp, bp + 16);
      acc = wmma_f16(af, bf, acc);
    }
    int nn = tn * 16 + (lane & 15);
#pragma unroll
    for (int r = 0; r < 8; ++r) {
      int a = tm * 16 + r + ((lane >> 4) * 8);
      float val;
      if (a < AG_) val = (nn < N_) ? acc[r] * 0.125f + bias1[((h * AG_) + a) * N_ + nn] : -1e30f;
      else         val = (nn < N_) ? 0.0f : -1e30f;
      sc[a * NPAD_ + nn] = (_Float16)val;
    }
  }
  __syncthreads();

  // ---- stage 1 softmax over n (rows = agents)
  if (tid < AGP_) {
    const _Float16* row = sc + tid * NPAD_;
    float mx = -1e30f;
    for (int n = 0; n < NPAD_; ++n) mx = fmaxf(mx, (float)row[n]);
    float sum = 0.f;
    for (int n = 0; n < NPAD_; ++n) sum += __expf((float)row[n] - mx);
    float inv = 1.f / sum;
    for (int n = 0; n < NPAD_; ++n) at[tid * NPAD_ + n] = (_Float16)(__expf((float)row[n] - mx) * inv);
  }
  __syncthreads();

  // ---- agent_v = attn1(64x224) . v(224x64) -> agvT[d][a]
  for (int t = wv; t < 16; t += 8) {
    int tm = t >> 2, tn = t & 3;
    v8f acc = {0,0,0,0,0,0,0,0};
#pragma unroll
    for (int k0 = 0; k0 < NPAD_; k0 += 32) {
      const _Float16* ap = at + (tm * 16 + (lane & 15)) * NPAD_ + k0 + ((lane >> 4) * 8);
      v16h af = ldfrag(ap, ap + 16);
      const _Float16* bp = vT + (tn * 16 + (lane & 15)) * NPAD_ + k0 + ((lane >> 4) * 8);
      v16h bf = ldfrag(bp, bp + 16);
      acc = wmma_f16(af, bf, acc);
    }
    int d = tn * 16 + (lane & 15);
#pragma unroll
    for (int r = 0; r < 8; ++r) {
      int a = tm * 16 + r + ((lane >> 4) * 8);
      agvT[d * AGP_ + a] = (_Float16)acc[r];
    }
  }
  __syncthreads();

  // ---- stage 2 scores: q(224x64) . ag^T -> sc[n][a] (224x64)
  for (int t = wv; t < 56; t += 8) {
    int tm = t >> 2, tn = t & 3;
    v8f acc = {0,0,0,0,0,0,0,0};
#pragma unroll
    for (int k0 = 0; k0 < HD_; k0 += 32) {
      const _Float16* ap = q_s + (tm * 16 + (lane & 15)) * HD_ + k0 + ((lane >> 4) * 8);
      v16h af = ldfrag(ap, ap + 16);
      const _Float16* bp = ag + (tn * 16 + (lane & 15)) * HD_ + k0 + ((lane >> 4) * 8);
      v16h bf = ldfrag(bp, bp + 16);
      acc = wmma_f16(af, bf, acc);
    }
    int a = tn * 16 + (lane & 15);
#pragma unroll
    for (int r = 0; r < 8; ++r) {
      int n = tm * 16 + r + ((lane >> 4) * 8);
      float val;
      if (a < AG_) val = (n < N_) ? acc[r] * 0.125f + bias2[((h * N_) + n) * AG_ + a] : 0.0f;
      else         val = -1e30f;
      sc[n * AGP_ + a] = (_Float16)val;
    }
  }
  __syncthreads();

  // ---- stage 2 softmax over a (rows = tokens)
  if (tid < NPAD_) {
    const _Float16* row = sc + tid * AGP_;
    float mx = -1e30f;
    for (int a = 0; a < AGP_; ++a) mx = fmaxf(mx, (float)row[a]);
    float sum = 0.f;
    for (int a = 0; a < AGP_; ++a) sum += __expf((float)row[a] - mx);
    float inv = 1.f / sum;
    for (int a = 0; a < AGP_; ++a) at[tid * AGP_ + a] = (_Float16)(__expf((float)row[a] - mx) * inv);
  }
  __syncthreads();

  // ---- out = attn2(224x64) . agent_v(64x64) + depthwise 3x3 conv on v
  for (int t = wv; t < 56; t += 8) {
    int tm = t >> 2, tn = t & 3;
    v8f acc = {0,0,0,0,0,0,0,0};
#pragma unroll
    for (int k0 = 0; k0 < AGP_; k0 += 32) {
      const _Float16* ap = at + (tm * 16 + (lane & 15)) * AGP_ + k0 + ((lane >> 4) * 8);
      v16h af = ldfrag(ap, ap + 16);
      const _Float16* bp = agvT + (tn * 16 + (lane & 15)) * AGP_ + k0 + ((lane >> 4) * 8);
      v16h bf = ldfrag(bp, bp + 16);
      acc = wmma_f16(af, bf, acc);
    }
    int d = tn * 16 + (lane & 15);
    int c = h * HD_ + d;
#pragma unroll
    for (int r = 0; r < 8; ++r) {
      int n = tm * 16 + r + ((lane >> 4) * 8);
      if (n < N_) {
        int y = n / 14, x = n - y * 14;
        float dv = dwcb[c];
#pragma unroll
        for (int ky = 0; ky < 3; ++ky) {
          int yy = y + ky - 1;
          if (yy < 0 || yy >= 14) continue;
#pragma unroll
          for (int kx = 0; kx < 3; ++kx) {
            int xx = x + kx - 1;
            if (xx < 0 || xx >= 14) continue;
            dv += dwcw[c * 9 + ky * 3 + kx] * (float)vT[d * NPAD_ + (yy * 14 + xx)];
          }
        }
        aout[((long)b * N_ + n) * C_ + c] = (_Float16)(acc[r] + dv);
      }
    }
  }
}

// ---------------- host launcher ----------------
extern "C" void kernel_launch(void* const* d_in, const int* in_sizes, int n_in,
                              void* d_out, int out_size, void* d_ws, size_t ws_size,
                              hipStream_t stream) {
  const float* x      = (const float*)d_in[0];
  const float* qkv_w  = (const float*)d_in[1];
  const float* qkv_b  = (const float*)d_in[2];
  const float* proj_w = (const float*)d_in[3];
  const float* proj_b = (const float*)d_in[4];
  const float* dwc_w  = (const float*)d_in[5];
  const float* dwc_b  = (const float*)d_in[6];
  const float* an_b   = (const float*)d_in[7];
  const float* na_b   = (const float*)d_in[8];
  const float* ah_b   = (const float*)d_in[9];
  const float* aw_b   = (const float*)d_in[10];
  const float* ha_b   = (const float*)d_in[11];
  const float* wa_b   = (const float*)d_in[12];
  float* out = (float*)d_out;

  const long M    = (long)B_ * N_;          // 100352
  const long XCNT = M * C_;                 // 51,380,224
  const long QKVC = M * 3 * C_;             // 154,140,672
  const int  BCNT = NH_ * AG_ * N_;         // 76,832

  char* p = (char*)d_ws;
  auto carve = [&](size_t bytes) {
    void* r = (void*)p;
    p += (bytes + 255) & ~(size_t)255;
    return r;
  };
  _Float16* x16   = (_Float16*)carve((size_t)XCNT * 2);
  _Float16* wq16  = (_Float16*)carve((size_t)(3 * C_) * C_ * 2);
  _Float16* wp16  = (_Float16*)carve((size_t)C_ * C_ * 2);
  _Float16* qkv16 = (_Float16*)carve((size_t)QKVC * 2);
  _Float16* ao16  = (_Float16*)carve((size_t)XCNT * 2);
  float*    bias1 = (float*)carve((size_t)BCNT * 4);
  float*    bias2 = (float*)carve((size_t)BCNT * 4);
  (void)in_sizes; (void)n_in; (void)out_size; (void)ws_size;

  cvt_f32_to_f16<<<(int)((XCNT + 255) / 256), 256, 0, stream>>>(x, x16, (int)XCNT);
  repack_weight<<<(3 * C_ * C_ + 255) / 256, 256, 0, stream>>>(qkv_w, wq16, 3 * C_, C_);
  repack_weight<<<(C_ * C_ + 255) / 256, 256, 0, stream>>>(proj_w, wp16, C_, C_);
  bias_kernel<<<(2 * BCNT + 255) / 256, 256, 0, stream>>>(an_b, na_b, ah_b, aw_b, ha_b, wa_b,
                                                          bias1, bias2);

  // qkv = x @ qkv_w^T + qkv_b  (f16 out)
  gemm_wmma<true><<<dim3((3 * C_) / 128, (int)(M / 128)), 256, 0, stream>>>(
      x16, wq16, qkv_b, (void*)qkv16, (int)M, C_, 3 * C_);

  // fused agent attention + dwc residual (f16 out), q/k tiles via Tensor Data Mover
  attn_kernel<<<B_ * NH_, 256, 0, stream>>>(qkv16, bias1, bias2, dwc_w, dwc_b, ao16);

  // out = attn_out @ proj_w^T + proj_b  (f32 out)
  gemm_wmma<false><<<dim3(C_ / 128, (int)(M / 128)), 256, 0, stream>>>(
      ao16, wp16, proj_b, (void*)out, (int)M, C_, C_);
}
